// FLAttention_19533511262507
// MI455X (gfx1250) — matrix operand order repaired and checked by reading, not compiled, not run on hardware
//
#include <hip/hip_runtime.h>
#include <hip/hip_bf16.h>

typedef __attribute__((ext_vector_type(16))) _Float16 v16h;
typedef __attribute__((ext_vector_type(8)))  float    v8f;

#define B_   64
#define D_   512
#define H_   8
#define QT   16     // q-tile per block
#define NW   4      // waves per block (wave32)
#define LN2_      0.6931471805599453f
#define EPS2_     (1e-8f * 0.6931471805599453f)   // eps pre-scaled by ln2
#define INV_SQRTH 0.35355339059327373f

__global__ __launch_bounds__(128)
void FLAttention_fused_wmma3_kernel(
    const float* __restrict__ x,
    const float* __restrict__ alpha_q,
    const float* __restrict__ alpha_k,
    const float* __restrict__ alpha_v,
    const float* __restrict__ beta_q,
    const float* __restrict__ beta_v,
    const float* __restrict__ sem_w,
    const float* __restrict__ sem_b,
    float* __restrict__ out)
{
    __shared__ float xs_s[D_];                   // transformed row of x (2 KB)
    __shared__ float qv_s[QT][H_];               // query values, pre-scaled by ln2
    __shared__ float m_s[QT][H_];                // softmax max * log2e
    __shared__ float part_s[2][H_][NW][QT];      // [num|den][h][wave][q] (4 KB)

    const int t     = threadIdx.x;
    const int wave  = t >> 5;
    const int lane  = t & 31;
    const int b     = blockIdx.y;
    const int qbase = blockIdx.x * QT;

    // ---- Phase 0: semantic-space affine transform of row b into LDS ----
    for (int i = t; i < D_; i += 128)
        xs_s[i] = x[b * D_ + i] * sem_w[i] + sem_b[i];
    __syncthreads();

    // ln2-scaled query values: qv' = (xs*aq + bq) * ln2
    if (t < QT * H_) {
        const int qi = t >> 3, h = t & 7;
        qv_s[qi][h] = (xs_s[qbase + qi] * alpha_q[h] + beta_q[h]) * LN2_;
    }
    __syncthreads();

    // ---- Phase A: softmax max via min |diff| — no transcendentals in loop ----
    // In ln2-scaled space: m2 = m*log2e = 1/(dmin' + eps')
    {
        const int qi = t >> 3, h = t & 7;
        const float qv2 = qv_s[qi][h];
        const float ak2 = alpha_k[h] * LN2_;
        float d0 = 3.4e38f, d1 = 3.4e38f, d2 = 3.4e38f, d3 = 3.4e38f;
        for (int k = 0; k < D_; k += 4) {
            d0 = fminf(d0, fabsf(fmaf(xs_s[k + 0], ak2, -qv2)));
            d1 = fminf(d1, fabsf(fmaf(xs_s[k + 1], ak2, -qv2)));
            d2 = fminf(d2, fabsf(fmaf(xs_s[k + 2], ak2, -qv2)));
            d3 = fminf(d3, fabsf(fmaf(xs_s[k + 3], ak2, -qv2)));
        }
        const float dmin = fminf(fminf(d0, d1), fminf(d2, d3));
        m_s[qi][h] = __builtin_amdgcn_rcpf(dmin + EPS2_);
    }
    __syncthreads();

    // ---- Phase B: per-head num/den reductions fused into WMMA ----
    // A (16x32 f16) = E[q,k] = exp2(rcp(d') - m2) for head h  (<= 1, f16-safe)
    // B (32x16 f16): col 0 = xs chunk, col 1 = ones, cols 2..15 = 0
    //   -> D col 0 = num_qh partial, D col 1 = den_qh partial (f32 accum)
    const int half = lane >> 4;   // K-half this lane covers
    const int row  = lane & 15;   // A: M (q-row); B/D: N (column)

    float qvr[H_], mr[H_], akr[H_];
#pragma unroll
    for (int h = 0; h < H_; ++h) {
        qvr[h] = qv_s[row][h];
        mr[h]  = m_s[row][h];
        akr[h] = alpha_k[h] * LN2_;
    }

    v8f acc[H_];
#pragma unroll
    for (int h = 0; h < H_; ++h)
        acc[h] = (v8f){0.f, 0.f, 0.f, 0.f, 0.f, 0.f, 0.f, 0.f};

    for (int c = wave; c < D_ / 32; c += NW) {   // 4 chunks of K=32 per wave
        const int kbase = c * 32;
        float xA[16];
        v16h  bb;
#pragma unroll
        for (int v = 0; v < 8; ++v) {
#pragma unroll
            for (int s = 0; s < 2; ++s) {
                const int j  = 2 * v + s;
                // A-matrix K stripe (ISA 7.12.2 16-bit A layout)
                const int kA = kbase + ((v < 4) ? (half * 8 + j)
                                                : (16 + half * 8 + (j - 8)));
                // B-matrix K stripe (same K order as A)
                const int kB = kA;
                xA[j] = xs_s[kA];
                const float xvB = xs_s[kB];
                bb[j] = (row == 0) ? (_Float16)xvB
                      : (row == 1) ? (_Float16)1.0f
                                   : (_Float16)0.0f;
            }
        }
#pragma unroll
        for (int h = 0; h < H_; ++h) {
            v16h a;
#pragma unroll
            for (int j = 0; j < 16; ++j) {
                // d' = (|xv*ak - qv| + eps) * ln2 ; E = exp2(1/d' - m2)
                const float d = fabsf(fmaf(xA[j], akr[h], -qvr[h])) + EPS2_;
                const float e = __builtin_amdgcn_exp2f(
                                    __builtin_amdgcn_rcpf(d) - mr[h]);
                a[j] = (_Float16)e;
            }
            acc[h] = __builtin_amdgcn_wmma_f32_16x16x32_f16(
                         false, a, false, bb, (short)0, acc[h], false, false);
        }
    }

    // ---- write num/den partials (D layout: VGPR r -> M = half*8 + r) ----
    if (row < 2) {
#pragma unroll
        for (int h = 0; h < H_; ++h)
#pragma unroll
            for (int r = 0; r < 8; ++r)
                part_s[row][h][wave][half * 8 + r] = acc[h][r];
    }
    __syncthreads();

    // ---- Final: reduce wave partials, normalize per head, add residual ----
    if (t < QT) {
        float combined = 0.f;
#pragma unroll
        for (int h = 0; h < H_; ++h) {
            float num = 0.f, den = 0.f;
#pragma unroll
            for (int w = 0; w < NW; ++w) {
                num += part_s[0][h][w][t];
                den += part_s[1][h][w][t];
            }
            combined += alpha_v[h] * num * __builtin_amdgcn_rcpf(den);
        }
        float cbeta = 0.f;
#pragma unroll
        for (int h = 0; h < H_; ++h) cbeta += beta_v[h];

        out[b * D_ + qbase + t] = xs_s[qbase + t] + (combined + cbeta) * INV_SQRTH;
    }
}

extern "C" void kernel_launch(void* const* d_in, const int* in_sizes, int n_in,
                              void* d_out, int out_size, void* d_ws, size_t ws_size,
                              hipStream_t stream) {
    const float* x       = (const float*)d_in[0];
    const float* alpha_q = (const float*)d_in[1];
    const float* alpha_k = (const float*)d_in[2];
    const float* alpha_v = (const float*)d_in[3];
    const float* beta_q  = (const float*)d_in[4];
    const float* beta_v  = (const float*)d_in[5];
    const float* sem_w   = (const float*)d_in[6];
    const float* sem_b   = (const float*)d_in[7];
    float* out = (float*)d_out;

    dim3 grid(D_ / QT, B_);   // (32, 64)
    FLAttention_fused_wmma3_kernel<<<grid, 128, 0, stream>>>(
        x, alpha_q, alpha_k, alpha_v, beta_q, beta_v, sem_w, sem_b, out);
}
